// HebbianBlock_35390530519119
// MI455X (gfx1250) — matrix operand order repaired and checked by reading, not compiled, not run on hardware
//
#include <hip/hip_runtime.h>

typedef __bf16 bf16;
typedef __bf16 v16bf __attribute__((ext_vector_type(16)));
typedef float  v8f   __attribute__((ext_vector_type(8)));
typedef unsigned int uint32x4 __attribute__((ext_vector_type(4)));
typedef int          int32x4  __attribute__((ext_vector_type(4)));
typedef int          int32x8  __attribute__((ext_vector_type(8)));

// ---------------------------------------------------------------------------
// WMMA helper: D = A(16x32 bf16) * B(32x16 bf16) + C(16x16 f32)
// ---------------------------------------------------------------------------
__device__ __forceinline__ v8f wmma_bf16(v16bf a, v16bf b, v8f c) {
  return __builtin_amdgcn_wmma_f32_16x16x32_bf16(
      /*neg_a=*/false, a, /*neg_b=*/false, b,
      /*c_mod=*/(short)0, c, /*reuse_a=*/false, /*reuse_b=*/false);
}

// ---------------------------------------------------------------------------
// Fragment loaders (wave32).
// A-matrix 16x32 bf16: lane l (row M = l&15), half h = l>>4 owns
//   K in [16h,16h+16) contiguous (same order as B).
// B-matrix 32x16 bf16: lane l (col N = l&15), half h owns K in [16h,16h+16)
//   contiguous.  B always loaded from the stored transpose ("row n of B^T").
// ---------------------------------------------------------------------------
__device__ __forceinline__ v16bf fragA(const bf16* row, int k0, int half) {
  v16bf f; const bf16* p = row + k0 + half * 16;
#pragma unroll
  for (int e = 0; e < 8; ++e) { f[e] = p[e]; f[e + 8] = p[e + 8]; }
  return f;
}
__device__ __forceinline__ v16bf fragA_s(const bf16* row, int k0, int half, float s) {
  v16bf f; const bf16* p = row + k0 + half * 16;
#pragma unroll
  for (int e = 0; e < 8; ++e) {
    f[e]     = (bf16)((float)p[e] * s);
    f[e + 8] = (bf16)((float)p[e + 8] * s);
  }
  return f;
}
__device__ __forceinline__ v16bf fragB(const bf16* rowBT, int k0, int half) {
  v16bf f; const bf16* p = rowBT + k0 + half * 16;
#pragma unroll
  for (int e = 0; e < 16; ++e) f[e] = p[e];
  return f;
}
__device__ __forceinline__ v16bf fragB_f(const float* rowBT, int k0, int half) {
  v16bf f; const float* p = rowBT + k0 + half * 16;
#pragma unroll
  for (int e = 0; e < 16; ++e) f[e] = (bf16)p[e];
  return f;
}
__device__ __forceinline__ v16bf fragB_ks(const bf16* rowBT, int k0, int half,
                                          const float* sc) {
  v16bf f; const int b0 = k0 + half * 16;
#pragma unroll
  for (int e = 0; e < 16; ++e) f[e] = (bf16)((float)rowBT[b0 + e] * sc[b0 + e]);
  return f;
}

// ---------------------------------------------------------------------------
// fp32 -> bf16 conversion (grid stride)
// ---------------------------------------------------------------------------
__global__ void cvt_kernel(const float* __restrict__ s, bf16* __restrict__ d, long n) {
  long i = (long)blockIdx.x * blockDim.x + threadIdx.x;
  long stride = (long)gridDim.x * blockDim.x;
  for (; i < n; i += stride) d[i] = (bf16)s[i];
}

// ---------------------------------------------------------------------------
// GEMM: Y(bf16, MxN) = A(bf16, MxK row-major) @ W^T  (W: NxK row-major)
// one wave -> 16x64 tile.  grid(M/16, N/256), block 128 (4 waves)
// ---------------------------------------------------------------------------
__global__ __launch_bounds__(128)
void gemm_bf16(const bf16* __restrict__ A, const bf16* __restrict__ W,
               bf16* __restrict__ Y, int M, int N, int K) {
  const int lane = threadIdx.x & 31, wv = threadIdx.x >> 5;
  const int half = lane >> 4, r15 = lane & 15;
  const int mt = blockIdx.x;
  const int nc0 = (blockIdx.y * 4 + wv) * 64;
  const bf16* arow = A + (long)(mt * 16 + r15) * K;
  const bf16* b0 = W + (long)(nc0 + r15) * K;
  const bf16* b1 = b0 + (long)16 * K;
  const bf16* b2 = b0 + (long)32 * K;
  const bf16* b3 = b0 + (long)48 * K;
  v8f a0 = {}, a1 = {}, a2 = {}, a3 = {};
  for (int k0 = 0; k0 < K; k0 += 32) {
    __builtin_prefetch(arow + k0 + 128, 0, 3);
    __builtin_prefetch(b0 + k0 + 128, 0, 3);
    v16bf a = fragA(arow, k0, half);
    a0 = wmma_bf16(a, fragB(b0, k0, half), a0);
    a1 = wmma_bf16(a, fragB(b1, k0, half), a1);
    a2 = wmma_bf16(a, fragB(b2, k0, half), a2);
    a3 = wmma_bf16(a, fragB(b3, k0, half), a3);
  }
#pragma unroll
  for (int i = 0; i < 8; ++i) {
    long row = (long)(mt * 16 + i + 8 * half);
    Y[row * N + nc0 + r15]      = (bf16)a0[i];
    Y[row * N + nc0 + 16 + r15] = (bf16)a1[i];
    Y[row * N + nc0 + 32 + r15] = (bf16)a2[i];
    Y[row * N + nc0 + 48 + r15] = (bf16)a3[i];
  }
}

// ---------------------------------------------------------------------------
// Final GEMM with residual: out_f32 = resid + OG(bf16) @ W_out^T
// ---------------------------------------------------------------------------
__global__ __launch_bounds__(128)
void gemm_out(const bf16* __restrict__ A, const bf16* __restrict__ W,
              const float* __restrict__ resid, float* __restrict__ Y,
              int M, int N, int K) {
  const int lane = threadIdx.x & 31, wv = threadIdx.x >> 5;
  const int half = lane >> 4, r15 = lane & 15;
  const int mt = blockIdx.x;
  const int nc0 = (blockIdx.y * 4 + wv) * 64;
  const bf16* arow = A + (long)(mt * 16 + r15) * K;
  const bf16* b0 = W + (long)(nc0 + r15) * K;
  const bf16* b1 = b0 + (long)16 * K;
  const bf16* b2 = b0 + (long)32 * K;
  const bf16* b3 = b0 + (long)48 * K;
  v8f a0 = {}, a1 = {}, a2 = {}, a3 = {};
  for (int k0 = 0; k0 < K; k0 += 32) {
    __builtin_prefetch(arow + k0 + 128, 0, 3);
    __builtin_prefetch(b0 + k0 + 128, 0, 3);
    v16bf a = fragA(arow, k0, half);
    a0 = wmma_bf16(a, fragB(b0, k0, half), a0);
    a1 = wmma_bf16(a, fragB(b1, k0, half), a1);
    a2 = wmma_bf16(a, fragB(b2, k0, half), a2);
    a3 = wmma_bf16(a, fragB(b3, k0, half), a3);
  }
#pragma unroll
  for (int i = 0; i < 8; ++i) {
    long row = (long)(mt * 16 + i + 8 * half);
    Y[row * N + nc0 + r15]      = resid[row * N + nc0 + r15]      + a0[i];
    Y[row * N + nc0 + 16 + r15] = resid[row * N + nc0 + 16 + r15] + a1[i];
    Y[row * N + nc0 + 32 + r15] = resid[row * N + nc0 + 32 + r15] + a2[i];
    Y[row * N + nc0 + 48 + r15] = resid[row * N + nc0 + 48 + r15] + a3[i];
  }
}

// ---------------------------------------------------------------------------
// Head projections: type 0 -> gate = sigmoid, 1 -> beta = sigmoid,
// 2 -> decay = -exp(A_log) * softplus(v + dt_bias).  Wsm = [Wg;Wb;Wa] (48xK).
// ---------------------------------------------------------------------------
__global__ __launch_bounds__(128)
void proj_small(const bf16* __restrict__ A, const bf16* __restrict__ Wsm,
                const float* __restrict__ Alog, const float* __restrict__ dtb,
                float* __restrict__ gate, float* __restrict__ beta,
                float* __restrict__ decay, int M, int K) {
  const int lane = threadIdx.x & 31, wv = threadIdx.x >> 5;
  const int half = lane >> 4, r15 = lane & 15;
  const int type = blockIdx.x;
  const int mt = blockIdx.y * 4 + wv;
  const bf16* arow = A + (long)(mt * 16 + r15) * K;
  const bf16* brow = Wsm + (long)(type * 16 + r15) * K;
  v8f acc = {};
  for (int k0 = 0; k0 < K; k0 += 32)
    acc = wmma_bf16(fragA(arow, k0, half), fragB(brow, k0, half), acc);
  const int hd = r15;
  const float al = Alog[hd], db = dtb[hd];
#pragma unroll
  for (int i = 0; i < 8; ++i) {
    long m = (long)(mt * 16 + i + 8 * half);
    float v = acc[i];
    if (type == 0)      gate[m * 16 + hd] = 1.f / (1.f + __expf(-v));
    else if (type == 1) beta[m * 16 + hd] = 1.f / (1.f + __expf(-v));
    else {
      float xx = v + db;
      float sp = (xx > 20.f) ? xx : log1pf(__expf(xx));
      decay[m * 16 + hd] = -__expf(al) * sp;
    }
  }
}

// ---------------------------------------------------------------------------
// Chunked delta-rule scan.  One 256-thread block (8 wave32) per (b,h).
// State S kept transposed in f32 LDS.  rk tile staged by the Tensor Data
// Mover (tensor_load_to_lds + s_wait_tensorcnt) when available.
// ---------------------------------------------------------------------------
__global__ __launch_bounds__(256)
void scan_kernel(const bf16* __restrict__ XB, const bf16* __restrict__ VB,
                 const float* __restrict__ gate, const float* __restrict__ beta,
                 const float* __restrict__ decay, bf16* __restrict__ OG, int Tt) {
  constexpr int LDB = 68;   // bf16 row stride: 64 data + 4 pad (=136B row)
  constexpr int LDT = 66;   // f32 state row stride (padded)
  __shared__ bf16 s_rk[64 * LDB];
  __shared__ bf16 s_wk[64 * LDB];
  __shared__ bf16 s_wkT[64 * LDB];
  __shared__ bf16 s_vbT[64 * LDB];
  __shared__ bf16 s_intra[64 * LDB];
  __shared__ float s_ST[64 * LDT];           // S^T : s_ST[e][d] = S[d][e]
  __shared__ float s_dec[64], s_eexp[64], s_escale[64], s_gate[64];

  const int tid = threadIdx.x, lane = tid & 31, wv = tid >> 5;
  const int half = lane >> 4, r15 = lane & 15;
  const int b = blockIdx.x >> 4, h = blockIdx.x & 15;

  for (int i = tid; i < 64 * LDT; i += 256) s_ST[i] = 0.f;
  __syncthreads();

  const int NCH = Tt / 64;
  const int row = tid >> 2, seg = (tid & 3) * 16;

  for (int n = 0; n < NCH; ++n) {
    const int t0 = n * 64;

    // ---- rk tile: TDM async load (64x64 bf16 tile, row stride 1024 elems,
    //      LDS pad 2 DWORDs per 32 DWORDs to land on LDB=68) ----
#if defined(__gfx1250__) && __has_builtin(__builtin_amdgcn_tensor_load_to_lds)
    if (wv == 0) {
      unsigned lds_base =
          (unsigned)(unsigned long long)(__attribute__((address_space(3))) bf16*)s_rk;
      unsigned long long ga =
          (unsigned long long)(XB + (long)(b * Tt + t0) * 1024 + h * 64);
      uint32x4 g0;
      g0[0] = 1u;                                   // count=1, no gather
      g0[1] = lds_base;                             // lds_addr
      g0[2] = (unsigned)(ga & 0xffffffffu);         // global_addr[31:0]
      g0[3] = (unsigned)((ga >> 32) & 0x01ffffffu)  // global_addr[56:32]
              | (2u << 30);                         // type = image(2)
      int32x8 g1;
      g1[0] = (int)((1u << 16)      // data_size = 2B
                  | (1u << 20)      // pad_enable
                  | (4u << 22)      // pad_interval: 32 DWORDs (=128B)
                  | (1u << 25));    // pad_amount: 2 DWORDs (=8B)
      g1[1] = (int)(64u << 16);     // tensor_dim0 = 64 (lo16 at bit48)
      g1[2] = (int)(64u << 16);     // dim0 hi=0 | tensor_dim1 = 64 (lo16)
      g1[3] = (int)(64u << 16);     // dim1 hi=0 | tile_dim0 = 64
      g1[4] = (int)64u;             // tile_dim1 = 64, tile_dim2 = 0
      g1[5] = (int)1024u;           // tensor_dim0_stride = 1024
      g1[6] = 0; g1[7] = 0;         // stride hi / tensor_dim1_stride = 0
      int32x4 z4 = {0, 0, 0, 0};
#if __clang_major__ >= 23
      int32x8 z8 = {0, 0, 0, 0, 0, 0, 0, 0};
      __builtin_amdgcn_tensor_load_to_lds(g0, g1, z4, z4, z8, 0);
#else
      __builtin_amdgcn_tensor_load_to_lds(g0, g1, z4, z4, 0);
#endif
      __builtin_amdgcn_s_wait_tensorcnt(0);
    }
#else
    {
      const bf16* xr = XB + (long)(b * Tt + t0 + row) * 1024 + h * 64 + seg;
#pragma unroll
      for (int j = 0; j < 16; ++j) s_rk[row * LDB + seg + j] = xr[j];
    }
#endif

    // ---- decay: load, cumsum, exponentials; gate row values ----
    if (tid < 64) s_dec[tid] = decay[(long)(b * Tt + t0 + tid) * 16 + h];
    __syncthreads();
    if (tid == 0) {
      float c = 0.f;
      for (int i = 0; i < 64; ++i) { c += s_dec[i]; s_dec[i] = c; }
    }
    __syncthreads();
    if (tid < 64) {
      s_eexp[tid]   = __expf(s_dec[tid]);
      s_escale[tid] = __expf(s_dec[63] - s_dec[tid]);
      s_gate[tid]   = gate[(long)(b * Tt + t0 + tid) * 16 + h];
    }

    // ---- stage wk (+transpose), v*beta (transposed) into LDS ----
    {
      const int tw = t0 + row - 1;          // token shift (zero at t==0)
      if (tw >= 0) {
        const bf16* wr = XB + (long)(b * Tt + tw) * 1024 + h * 64 + seg;
#pragma unroll
        for (int j = 0; j < 16; ++j) {
          bf16 w = wr[j];
          s_wk[row * LDB + seg + j] = w;
          s_wkT[(seg + j) * LDB + row] = w;
        }
      } else {
#pragma unroll
        for (int j = 0; j < 16; ++j) {
          s_wk[row * LDB + seg + j] = (bf16)0.f;
          s_wkT[(seg + j) * LDB + row] = (bf16)0.f;
        }
      }
      const float bet = beta[(long)(b * Tt + t0 + row) * 16 + h];
      const bf16* vr = VB + (long)(b * Tt + t0 + row) * 1024 + h * 64 + seg;
#pragma unroll
      for (int j = 0; j < 16; ++j)
        s_vbT[(seg + j) * LDB + row] = (bf16)((float)vr[j] * bet);
    }
    __syncthreads();

    // ---- matmul 1: intra = strict_lower((rk @ wk^T) * L) ----
#pragma unroll
    for (int s = 0; s < 2; ++s) {
      const int t = wv * 2 + s, ti = t >> 2, tj = t & 3;
      const bf16* ar = s_rk + (ti * 16 + r15) * LDB;
      const bf16* br = s_wk + (tj * 16 + r15) * LDB;   // (wk^T)^T = wk rows
      v8f acc = {};
      acc = wmma_bf16(fragA(ar, 0, half),  fragB(br, 0, half),  acc);
      acc = wmma_bf16(fragA(ar, 32, half), fragB(br, 32, half), acc);
#pragma unroll
      for (int i = 0; i < 8; ++i) {
        const int r = ti * 16 + i + 8 * half, cc = tj * 16 + r15;
        const float val = (r > cc) ? acc[i] * __expf(s_dec[r] - s_dec[cc]) : 0.f;
        s_intra[r * LDB + cc] = (bf16)val;
      }
    }
    __syncthreads();

    // ---- matmul 2: o = (rk * e^dec) @ S + intra @ vb; gate; store ----
#pragma unroll
    for (int s = 0; s < 2; ++s) {
      const int t = wv * 2 + s, ti = t >> 2, tj = t & 3;
      v8f acc = {};
      {
        const float sr = s_eexp[ti * 16 + r15];
        const bf16* ar = s_rk + (ti * 16 + r15) * LDB;
        const float* br = s_ST + (tj * 16 + r15) * LDT;  // col e of S = row e of S^T
        acc = wmma_bf16(fragA_s(ar, 0, half, sr),  fragB_f(br, 0, half),  acc);
        acc = wmma_bf16(fragA_s(ar, 32, half, sr), fragB_f(br, 32, half), acc);
      }
      {
        const bf16* ar = s_intra + (ti * 16 + r15) * LDB;
        const bf16* br = s_vbT + (tj * 16 + r15) * LDB;  // col e of vb = row e of vb^T
        acc = wmma_bf16(fragA(ar, 0, half),  fragB(br, 0, half),  acc);
        acc = wmma_bf16(fragA(ar, 32, half), fragB(br, 32, half), acc);
      }
#pragma unroll
      for (int i = 0; i < 8; ++i) {
        const int r = ti * 16 + i + 8 * half, e = tj * 16 + r15;
        OG[(long)(b * Tt + t0 + r) * 1024 + h * 64 + e] = (bf16)(acc[i] * s_gate[r]);
      }
    }
    __syncthreads();

    // ---- matmul 3: S^T = S^T * e^{dec_last} + vb^T @ (wk * escale) ----
    const float gL = __expf(s_dec[63]);
#pragma unroll
    for (int s = 0; s < 2; ++s) {
      const int t = wv * 2 + s, ti = t >> 2, tj = t & 3;
      const bf16* ar = s_vbT + (ti * 16 + r15) * LDB;
      const bf16* br = s_wkT + (tj * 16 + r15) * LDB;  // col d of wk' = row d of wk^T
      v8f acc = {};
      acc = wmma_bf16(fragA(ar, 0, half),  fragB_ks(br, 0, half, s_escale),  acc);
      acc = wmma_bf16(fragA(ar, 32, half), fragB_ks(br, 32, half, s_escale), acc);
#pragma unroll
      for (int i = 0; i < 8; ++i) {
        const int e = ti * 16 + i + 8 * half, d = tj * 16 + r15;
        s_ST[e * LDT + d] = s_ST[e * LDT + d] * gL + acc[i];
      }
    }
    __syncthreads();
  }
}

// ---------------------------------------------------------------------------
// launcher
// ---------------------------------------------------------------------------
extern "C" void kernel_launch(void* const* d_in, const int* in_sizes, int n_in,
                              void* d_out, int out_size, void* d_ws, size_t ws_size,
                              hipStream_t stream) {
  (void)in_sizes; (void)n_in; (void)out_size; (void)ws_size;
  constexpr int Bb = 4, Tt = 4096, Dm = 1024, Hh = 16;
  constexpr int M = Bb * Tt, K = Dm, N = Dm;

  const float* x    = (const float*)d_in[0];
  const float* Wwr  = (const float*)d_in[1];
  const float* Wg   = (const float*)d_in[2];
  const float* Wb   = (const float*)d_in[3];
  const float* Wa   = (const float*)d_in[4];
  const float* Wo   = (const float*)d_in[5];
  const float* Alog = (const float*)d_in[6];
  const float* dtb  = (const float*)d_in[7];
  float* y = (float*)d_out;

  char* ws = (char*)d_ws;
  size_t off = 0;
  auto alloc = [&](size_t bytes) -> void* {
    void* p = ws + off;
    off += (bytes + 255) & ~(size_t)255;
    return p;
  };
  bf16* xb   = (bf16*)alloc((size_t)M * K * 2);   // x in bf16
  bf16* vb   = (bf16*)alloc((size_t)M * K * 2);   // v = x @ W_write^T (bf16)
  bf16* og   = (bf16*)alloc((size_t)M * K * 2);   // gated o (bf16)
  bf16* wwb  = (bf16*)alloc((size_t)N * K * 2);
  bf16* wob  = (bf16*)alloc((size_t)N * K * 2);
  bf16* wsm  = (bf16*)alloc((size_t)48 * K * 2);
  float* gate  = (float*)alloc((size_t)M * Hh * 4);
  float* beta  = (float*)alloc((size_t)M * Hh * 4);
  float* decay = (float*)alloc((size_t)M * Hh * 4);

  // 1) bf16 conversions
  cvt_kernel<<<4096, 256, 0, stream>>>(x, xb, (long)M * K);
  cvt_kernel<<<1024, 256, 0, stream>>>(Wwr, wwb, (long)N * K);
  cvt_kernel<<<1024, 256, 0, stream>>>(Wo, wob, (long)N * K);
  cvt_kernel<<<64, 256, 0, stream>>>(Wg, wsm, (long)16 * K);
  cvt_kernel<<<64, 256, 0, stream>>>(Wb, wsm + 16 * K, (long)16 * K);
  cvt_kernel<<<64, 256, 0, stream>>>(Wa, wsm + 32 * K, (long)16 * K);

  // 2) v = x @ W_write^T
  gemm_bf16<<<dim3(M / 16, N / 256), 128, 0, stream>>>(xb, wwb, vb, M, N, K);

  // 3) gate / beta / decay head projections
  proj_small<<<dim3(3, M / 64), 128, 0, stream>>>(xb, wsm, Alog, dtb,
                                                  gate, beta, decay, M, K);

  // 4) chunked scan -> gated o (bf16)
  scan_kernel<<<Bb * Hh, 256, 0, stream>>>(xb, vb, gate, beta, decay, og, Tt);

  // 5) y = x + og @ W_out^T
  gemm_out<<<dim3(M / 16, N / 256), 128, 0, stream>>>(og, wob, x, y, M, N, K);
}
